// GATLayer_29712583753905
// MI455X (gfx1250) — matrix operand order repaired and checked by reading, not compiled, not run on hardware
//
#include <hip/hip_runtime.h>

typedef __attribute__((ext_vector_type(16))) _Float16 v16h;
typedef __attribute__((ext_vector_type(8)))  _Float16 v8h;
typedef __attribute__((ext_vector_type(4)))  _Float16 v4h;
typedef __attribute__((ext_vector_type(8)))  float    v8f;
typedef __attribute__((ext_vector_type(4)))  int      v4i;

#define NODES  4096
#define NF     512      // HEADS*OUT_F
#define NBPAD  640      // padded column count for main GEMM (512 num + 8 den + pad)
#define KT     32       // K-step per WMMA iteration
#define SA     40       // LDS row stride (halfs), padded: 80B = 20 banks -> conflict-free

// Async global->LDS DMA path (CDNA5): guarded, falls back to VGPR staging.
#if defined(__has_builtin)
#if __has_builtin(__builtin_amdgcn_global_load_async_to_lds_b128) && \
    __has_builtin(__builtin_amdgcn_s_wait_asynccnt)
#define ASYNC_LDS 1
#endif
#endif
#ifndef ASYNC_LDS
#define ASYNC_LDS 0
#endif

#if ASYNC_LDS
typedef __attribute__((address_space(1))) v4i GV4i;   // global int4*
typedef __attribute__((address_space(3))) v4i SV4i;   // LDS int4*
__device__ __forceinline__ void async_copy_b128(const _Float16* g, _Float16* l) {
  __builtin_amdgcn_global_load_async_to_lds_b128((GV4i*)(g), (SV4i*)(l), 0, 0);
}
#endif

// ---------------- workspace layout (bytes) ----------------
// hfeat f32 [4096][512]  @ 0          (8,388,608)
// wbuf  f32 [4096][8]    @ 8,388,608  (131,072)
// Bt    f16 [640][4096]  @ 8,519,680  (5,242,880)   (K-major B for main GEMM)
// Wt    f16 [512][256]   @ 13,762,560 (262,144)     (K-major B for GEMM1)
// Cws   f32 [4096][640]  @ 14,024,704 (10,485,760)
// total 24,510,464 bytes
#define WS_HFEAT 0
#define WS_WBUF  8388608
#define WS_BT    8519680
#define WS_WT    13762560
#define WS_CWS   14024704

// ---- element converters for the A-tile global->LDS path ----
struct CvtF32 {
  typedef float  T;
  typedef float4 V;
  static __device__ inline _Float16 cvt(float v) { return (_Float16)v; }
};
struct CvtAdj {
  typedef int  T;
  typedef int4 V;
  static __device__ inline _Float16 cvt(int v) { return v != 0 ? (_Float16)1.0f : (_Float16)0.0f; }
};

// Tiled WMMA GEMM:  C[M][ldc] = A[M][K] * Bt[n][K]^T   (Bt is B transposed, K-major)
// Workgroup: 256 threads = 8 waves, tile 128(M) x 128(N).
// Wave: 64(M) x 32(N) = 4x2 WMMA 16x16 tiles, 8 v_wmma per 32-K step.
// Double-buffered LDS; B tile streamed with GLOBAL_LOAD_ASYNC_TO_LDS (ASYNCcnt),
// A tile staged through VGPRs (needs int32/f32 -> f16 conversion).
template <typename CVT>
__global__ __launch_bounds__(256, 1)
void wmma_gemm(const typename CVT::T* __restrict__ A,
               const _Float16* __restrict__ Bt,
               float* __restrict__ C,
               int K, int ldc)
{
  __shared__ _Float16 As[2][128 * SA];
  __shared__ _Float16 Bs[2][128 * SA];

  const int tid  = threadIdx.x;
  const int lane = tid & 31;
  const int wave = tid >> 5;
  const int lrow = lane & 15;   // M/N index within a 16-wide tile
  const int lhi  = lane >> 4;   // half-wave selector
  const int mw   = (wave & 1) * 64;
  const int nw   = (wave >> 1) * 32;
  const long iBase = (long)blockIdx.x * 128;
  const long nBase = (long)blockIdx.y * 128;

  v8f acc[4][2] = {};

  typename CVT::V va[4];
#if !ASYNC_LDS
  uint4 vb[2];
#endif

  // ---- prologue: fetch tile 0 and stage into LDS buffer 0 ----
#if ASYNC_LDS
  #pragma unroll
  for (int l = 0; l < 2; ++l) {
    int idx = tid + 256 * l;
    int n = idx >> 2, seg = idx & 3;
    async_copy_b128(Bt + (nBase + n) * (long)K + seg * 8, &Bs[0][n * SA + seg * 8]);
  }
#endif
  #pragma unroll
  for (int l = 0; l < 4; ++l) {
    int idx = tid + 256 * l;
    int row = idx >> 3, seg = idx & 7;
    va[l] = *(const typename CVT::V*)(A + (iBase + row) * (long)K + seg * 4);
  }
#if !ASYNC_LDS
  #pragma unroll
  for (int l = 0; l < 2; ++l) {
    int idx = tid + 256 * l;
    int n = idx >> 2, seg = idx & 3;
    vb[l] = *(const uint4*)(Bt + (nBase + n) * (long)K + seg * 8);
  }
#endif
  #pragma unroll
  for (int l = 0; l < 4; ++l) {
    int idx = tid + 256 * l;
    int row = idx >> 3, seg = idx & 7;
    v4h h;
    h.x = CVT::cvt(va[l].x); h.y = CVT::cvt(va[l].y);
    h.z = CVT::cvt(va[l].z); h.w = CVT::cvt(va[l].w);
    *(v4h*)(&As[0][row * SA + seg * 4]) = h;
  }
#if ASYNC_LDS
  __builtin_amdgcn_s_wait_asynccnt(0);
#else
  #pragma unroll
  for (int l = 0; l < 2; ++l) {
    int idx = tid + 256 * l;
    int n = idx >> 2, seg = idx & 3;
    *(uint4*)(&Bs[0][n * SA + seg * 8]) = vb[l];
  }
#endif
  __syncthreads();

  const int nk = K / KT;
  for (int kt = 0; kt < nk; ++kt) {
    const int  cur  = kt & 1;
    const int  nxt  = cur ^ 1;
    const bool more = (kt + 1) < nk;
    const int  k1   = (kt + 1) * KT;

    // ---- issue next tile's loads early (latency hidden by WMMAs) ----
    if (more) {
#if ASYNC_LDS
      // B tile: direct async DMA into the other LDS buffer, no VGPRs
      #pragma unroll
      for (int l = 0; l < 2; ++l) {
        int idx = tid + 256 * l;
        int n = idx >> 2, seg = idx & 3;
        async_copy_b128(Bt + (nBase + n) * (long)K + k1 + seg * 8,
                        &Bs[nxt][n * SA + seg * 8]);
      }
#endif
      #pragma unroll
      for (int l = 0; l < 4; ++l) {
        int idx = tid + 256 * l;
        int row = idx >> 3, seg = idx & 7;
        va[l] = *(const typename CVT::V*)(A + (iBase + row) * (long)K + k1 + seg * 4);
      }
#if !ASYNC_LDS
      #pragma unroll
      for (int l = 0; l < 2; ++l) {
        int idx = tid + 256 * l;
        int n = idx >> 2, seg = idx & 3;
        vb[l] = *(const uint4*)(Bt + (nBase + n) * (long)K + k1 + seg * 8);
      }
#endif
      if (kt + 2 < nk) {  // warm L2 two tiles ahead on the A stream
        __builtin_prefetch(A + (iBase + (tid >> 1)) * (long)K + (kt + 2) * KT, 0, 3);
      }
    }

    // ---- compute on current buffer ----
    // B fragments: lane holds col N=nw+bi*16+lrow, K = t + 16*lhi (contiguous)
    v16h bf[2];
    #pragma unroll
    for (int bi = 0; bi < 2; ++bi) {
      int n  = nw + bi * 16 + lrow;
      int kb = lhi * 16;
      v8h lo = *(const v8h*)(&Bs[cur][n * SA + kb]);
      v8h hi = *(const v8h*)(&Bs[cur][n * SA + kb + 8]);
      bf[bi] = __builtin_shufflevector(lo, hi, 0,1,2,3,4,5,6,7,8,9,10,11,12,13,14,15);
    }
    // A fragments: lane holds row M=mw+ai*16+lrow, K = t + 16*lhi (contiguous)
    #pragma unroll
    for (int ai = 0; ai < 4; ++ai) {
      int m  = mw + ai * 16 + lrow;
      int ka = lhi * 16;
      v8h lo = *(const v8h*)(&As[cur][m * SA + ka]);
      v8h hi = *(const v8h*)(&As[cur][m * SA + ka + 8]);
      v16h af = __builtin_shufflevector(lo, hi, 0,1,2,3,4,5,6,7,8,9,10,11,12,13,14,15);
      #pragma unroll
      for (int bi = 0; bi < 2; ++bi) {
        acc[ai][bi] = __builtin_amdgcn_wmma_f32_16x16x32_f16(
            false, af, false, bf[bi], (short)0, acc[ai][bi], false, false);
      }
    }

    // ---- drain prefetched A tile into the other LDS buffer ----
    if (more) {
      #pragma unroll
      for (int l = 0; l < 4; ++l) {
        int idx = tid + 256 * l;
        int row = idx >> 3, seg = idx & 7;
        v4h h;
        h.x = CVT::cvt(va[l].x); h.y = CVT::cvt(va[l].y);
        h.z = CVT::cvt(va[l].z); h.w = CVT::cvt(va[l].w);
        *(v4h*)(&As[nxt][row * SA + seg * 4]) = h;
      }
#if !ASYNC_LDS
      #pragma unroll
      for (int l = 0; l < 2; ++l) {
        int idx = tid + 256 * l;
        int n = idx >> 2, seg = idx & 3;
        *(uint4*)(&Bs[nxt][n * SA + seg * 8]) = vb[l];
      }
#endif
    }
#if ASYNC_LDS
    if (more) __builtin_amdgcn_s_wait_asynccnt(0);
#endif
    __syncthreads();
  }

  // ---- epilogue: D VGPR r -> row M = base + r + 8*lhi, col N = base + lrow ----
  #pragma unroll
  for (int ai = 0; ai < 4; ++ai) {
    #pragma unroll
    for (int bi = 0; bi < 2; ++bi) {
      long gm = iBase + mw + ai * 16 + lhi * 8;
      long gn = nBase + nw + bi * 16 + lrow;
      #pragma unroll
      for (int r = 0; r < 8; ++r)
        C[(gm + r) * (long)ldc + gn] = acc[ai][bi][r];
    }
  }
}

// Wt[n][k] = (f16) W[k][n]   (512 x 256)
__global__ __launch_bounds__(256)
void wt_build(const float* __restrict__ W, _Float16* __restrict__ Wt) {
  int t = blockIdx.x * 256 + threadIdx.x;   // t = n*256 + k
  int n = t >> 8;
  int k = t & 255;
  Wt[t] = (_Float16)W[k * 512 + n];
}

// wbuf[j][h] = exp( sum_d hfeat[j][h*64+d] * a_dst[h][d] )   (s_src cancels in softmax)
__global__ __launch_bounds__(256)
void sdst_exp(const float* __restrict__ hfeat, const float* __restrict__ attn,
              float* __restrict__ wbuf) {
  int t = blockIdx.x * 256 + threadIdx.x;   // t = j*8 + h
  int j = t >> 3;
  int h = t & 7;
  const float* hp = hfeat + (long)j * NF + h * 64;
  const float* ap = attn + h * 128 + 64;    // a_dst = attn[0, h, 64:128]
  float s = 0.f;
  #pragma unroll 8
  for (int d = 0; d < 64; ++d) s += hp[d] * ap[d];
  wbuf[t] = expf(s);
}

// Bt[n][j] (K-major):  n<512 -> w[j][n>>6]*h[j][n] ; 512<=n<520 -> w[j][n-512] ; else 0
__global__ __launch_bounds__(256)
void bt_build(const float* __restrict__ hfeat, const float* __restrict__ wbuf,
              _Float16* __restrict__ Bt) {
  long idx = (long)blockIdx.x * 256 + threadIdx.x;  // idx = n*4096 + j
  int n = (int)(idx >> 12);
  int j = (int)(idx & 4095);
  float v;
  if (n < 512)      v = wbuf[j * 8 + (n >> 6)] * hfeat[(long)j * NF + n];
  else if (n < 520) v = wbuf[j * 8 + (n - 512)];
  else              v = 0.f;
  Bt[idx] = (_Float16)v;
}

// out[i][c] = num / den ,  den at Cws col 512 + (c>>6)
__global__ __launch_bounds__(256)
void finalize(const float* __restrict__ Cws, float* __restrict__ out) {
  int t = blockIdx.x * 256 + threadIdx.x;   // t = i*512 + c
  int i = t >> 9;
  int c = t & 511;
  float num = Cws[(long)i * NBPAD + c];
  float den = Cws[(long)i * NBPAD + 512 + (c >> 6)];
  out[t] = num / den;
}

extern "C" void kernel_launch(void* const* d_in, const int* in_sizes, int n_in,
                              void* d_out, int out_size, void* d_ws, size_t ws_size,
                              hipStream_t stream) {
  (void)in_sizes; (void)n_in; (void)out_size; (void)ws_size;
  const float* x    = (const float*)d_in[0];
  const int*   adj  = (const int*)d_in[1];
  const float* W    = (const float*)d_in[2];
  const float* attn = (const float*)d_in[3];
  float* out = (float*)d_out;

  char* ws = (char*)d_ws;
  float*    hfeat = (float*)(ws + WS_HFEAT);
  float*    wbuf  = (float*)(ws + WS_WBUF);
  _Float16* Bt    = (_Float16*)(ws + WS_BT);
  _Float16* Wt    = (_Float16*)(ws + WS_WT);
  float*    Cws   = (float*)(ws + WS_CWS);

  // 1) Wt = W^T (f16)
  wt_build<<<512, 256, 0, stream>>>(W, Wt);
  // 2) hfeat = x @ W   (M=4096, K=256, N=512)
  wmma_gemm<CvtF32><<<dim3(32, 4), 256, 0, stream>>>(x, Wt, hfeat, 256, NF);
  // 3) w = exp(s_dst)
  sdst_exp<<<128, 256, 0, stream>>>(hfeat, attn, wbuf);
  // 4) build fused B = [w*h | w | 0]  (K-major, 640 x 4096)
  bt_build<<<10240, 256, 0, stream>>>(hfeat, wbuf, Bt);
  // 5) Cws = f16(adj) @ B   (M=4096, K=4096, N=640)
  wmma_gemm<CvtAdj><<<dim3(32, 5), 256, 0, stream>>>(adj, Bt, Cws, NODES, NBPAD);
  // 6) out = num / den
  finalize<<<8192, 256, 0, stream>>>(Cws, out);
}
